// GemmaAttention_79542794322413
// MI455X (gfx1250) — matrix operand order repaired and checked by reading, not compiled, not run on hardware
//
#include <hip/hip_runtime.h>
#include <cstdint>
#include <cstddef>

// ---------------- types & constants ----------------
typedef __attribute__((ext_vector_type(16))) __bf16 v16bf;
typedef __attribute__((ext_vector_type(8)))  float  v8f;

#define S_LEN   2048
#define HID     3072
#define NH      16
#define NKV     8
#define HD      256
#define QDIM    4096   // NH*HD
#define KVDIM   2048   // NKV*HD
#define SCALING 0.0625f
#define SOFTCAP 50.0f
#define NEG_INF (-2.3819763e+38f)

union FragU {
    v16bf v;
    __bf16 h[16];
    unsigned u[8];
    uint4 q[2];
};

__device__ inline __bf16 f2bf(float f) { return (__bf16)f; }  // hw fptrunc (RNE)

__device__ inline void store4bf(__bf16* dst, float4 f) {      // one 8B LDS store
    __align__(8) __bf16 tmp[4] = { (__bf16)f.x, (__bf16)f.y, (__bf16)f.z, (__bf16)f.w };
    *(uint2*)dst = *(const uint2*)tmp;
}

__device__ inline v8f vzero8() {
    v8f z;
#pragma unroll
    for (int i = 0; i < 8; ++i) z[i] = 0.0f;
    return z;
}

__device__ inline float fast_tanh(float x) {                  // v_exp + v_rcp path
    x = fminf(fmaxf(x, -10.0f), 10.0f);
    float e = __expf(2.0f * x);
    return (e - 1.0f) / (e + 1.0f);
}

// ---------------- WMMA fragment access ----------------
// Swizzled A staging: tile-major [tile][lane][16 halves]; each lane's whole
// 16x32 A-fragment is 32 contiguous bytes -> 2x ds_load_b128.
// Element (m,k) of a 16x32 tile maps to lane = m + 16*((k>>3)&1),
// half = (k&7) + 8*(k>>4).
__device__ inline v16bf load_A_swz(const __bf16* base, int tile, int lane) {
    const uint4* p = (const uint4*)(base + ((size_t)((tile << 5) + lane) << 4));
    FragU r;
    r.q[0] = p[0];
    r.q[1] = p[1];
    return r.v;
}

// B-matrix 32x16 bf16 row-major in LDS: lane L holds column N=L%16, half s = K 8*(L/16)+(s&7)+16*(s>>3)
__device__ inline v16bf load_B32x16(const __bf16* tile, int ld, int lane) {
    const int n = lane & 15, hb = (lane >> 4) * 8;
    FragU r;
#pragma unroll
    for (int s = 0; s < 16; ++s)
        r.h[s] = tile[(size_t)(hb + (s & 7) + 16 * (s >> 3)) * ld + n];
    return r.v;
}

__device__ inline v8f wmma_bf16(v16bf a, v16bf b, v8f c) {
    return __builtin_amdgcn_wmma_f32_16x16x32_bf16(
        false, a, false, b, (short)0, c, false, false);
}

// ---------------- 128x128 block GEMM, double-buffered ----------------
__device__ void gemm_block_128(const float* __restrict__ A, int lda,
                               const float* __restrict__ B, int ldb,
                               float* __restrict__ C, int ldc,
                               int K, int row0, int col0)
{
    __shared__ __align__(16) __bf16 Asw[2][8 * 32 * 16];  // 8 tiles, swizzled
    __shared__ __align__(16) __bf16 Bs[2][32][128];

    const int t    = threadIdx.x;
    const int lane = t & 31;
    const int wid  = t >> 5;
    const int wm   = (wid >> 2) * 64;   // wave rows (4 M-tiles)
    const int wn   = (wid & 3) * 32;    // wave cols (2 N-tiles)
    const int ti0  = (wid >> 2) * 4;    // first A tile index

    v8f acc[4][2];
#pragma unroll
    for (int i = 0; i < 4; ++i)
#pragma unroll
        for (int j = 0; j < 2; ++j) acc[i][j] = vzero8();

    float4 ra[4], rb[4];
    auto load_regs = [&](int k0) {
#pragma unroll
        for (int i = 0; i < 4; ++i) {
            int idx = i * 256 + t;
            ra[i] = *(const float4*)(A + (size_t)(row0 + (idx >> 3)) * lda
                                       + k0 + ((idx & 7) << 2));
            rb[i] = *(const float4*)(B + (size_t)(k0 + (idx >> 5)) * ldb
                                       + col0 + ((idx & 31) << 2));
        }
    };
    auto store_lds = [&](int bufi) {
#pragma unroll
        for (int i = 0; i < 4; ++i) {
            int idx = i * 256 + t;
            // A swizzled store
            int r = idx >> 3, c = (idx & 7) << 2;
            int tile = r >> 4;
            int L = (r & 15) + 16 * ((c >> 3) & 1);
            int h0 = (c & 7) + 8 * (c >> 4);
            store4bf(&Asw[bufi][((tile << 5) + L) * 16 + h0], ra[i]);
            // B row-major store
            int br = idx >> 5, bc = (idx & 31) << 2;
            store4bf(&Bs[bufi][br][bc], rb[i]);
        }
    };

    load_regs(0);
    store_lds(0);
    __syncthreads();

    int buf = 0;
    for (int k0 = 0; k0 < K; k0 += 32) {
        const bool has_next = (k0 + 32 < K);
        if (has_next) load_regs(k0 + 32);   // overlap next slab with WMMAs

        v16bf bfr[2];
#pragma unroll
        for (int j = 0; j < 2; ++j)
            bfr[j] = load_B32x16(&Bs[buf][0][wn + 16 * j], 128, lane);
#pragma unroll
        for (int i = 0; i < 4; ++i) {
            v16bf a = load_A_swz(Asw[buf], ti0 + i, lane);
#pragma unroll
            for (int j = 0; j < 2; ++j)
                acc[i][j] = wmma_bf16(a, bfr[j], acc[i][j]);
        }

        if (has_next) store_lds(buf ^ 1);
        __syncthreads();
        buf ^= 1;
    }

    const int g = lane >> 4;
    const int n = lane & 15;
#pragma unroll
    for (int i = 0; i < 4; ++i)
#pragma unroll
        for (int j = 0; j < 2; ++j) {
            float* cp = C + (size_t)(row0 + wm + 16 * i + 8 * g) * ldc
                          + col0 + wn + 16 * j + n;
#pragma unroll
            for (int v = 0; v < 8; ++v) cp[(size_t)v * ldc] = acc[i][j][v];
        }
}

// ---------------- kernel 1: fused QKV projection ----------------
__global__ __launch_bounds__(256) void qkv_gemm_kernel(
    const float* __restrict__ X,
    const float* __restrict__ wq, const float* __restrict__ wk,
    const float* __restrict__ wv,
    float* __restrict__ xq, float* __restrict__ xk, float* __restrict__ xv)
{
    int cb   = blockIdx.x * 128;
    int row0 = blockIdx.y * 128;
    const float* B; float* C; int ldb; int col0;
    if (cb < QDIM)              { B = wq; C = xq; ldb = QDIM;  col0 = cb; }
    else if (cb < QDIM + KVDIM) { B = wk; C = xk; ldb = KVDIM; col0 = cb - QDIM; }
    else                        { B = wv; C = xv; ldb = KVDIM; col0 = cb - QDIM - KVDIM; }
    gemm_block_128(X, HID, B, ldb, C, ldb, HID, row0, col0);
}

// ---------------- kernel 2: RMSNorm + RoPE + bf16 cast ----------------
__global__ __launch_bounds__(256) void norm_rope_kernel(
    const float* __restrict__ xq, const float* __restrict__ xk,
    const float* __restrict__ xv,
    const float* __restrict__ cosb, const float* __restrict__ sinb,
    const float* __restrict__ qw, const float* __restrict__ kw,
    __bf16* __restrict__ qb, __bf16* __restrict__ kb, __bf16* __restrict__ vb)
{
    const int s = blockIdx.x;
    const int y = blockIdx.y;     // 0..15 q, 16..23 k, 24..31 v
    const int d = threadIdx.x;
    __shared__ float red[256];
    __shared__ float nrm[256];

    const float* src; __bf16* dst; const float* w;
    if (y < NH) {
        src = xq + ((size_t)s * NH + y) * HD;
        dst = qb + ((size_t)s * NH + y) * HD;
        w   = qw;
    } else if (y < NH + NKV) {
        int h = y - NH;
        src = xk + ((size_t)s * NKV + h) * HD;
        dst = kb + ((size_t)s * NKV + h) * HD;
        w   = kw;
    } else {
        int h = y - NH - NKV;
        const float* vs = xv + ((size_t)s * NKV + h) * HD;
        vb[((size_t)s * NKV + h) * HD + d] = f2bf(vs[d]);
        return;
    }

    float x = src[d];
    red[d] = x * x;
    __syncthreads();
#pragma unroll
    for (int o = 128; o > 0; o >>= 1) {
        if (d < o) red[d] += red[d + o];
        __syncthreads();
    }
    float mean = red[0] * (1.0f / HD);
    float nv = x * rsqrtf(mean + 1e-6f) * (1.0f + w[d]);
    nrm[d] = nv;
    __syncthreads();

    float out;
    if (d < 128) {
        float c = cosb[(size_t)s * 128 + d], sn = sinb[(size_t)s * 128 + d];
        out = nrm[d] * c - nrm[d + 128] * sn;
    } else {
        int dd = d - 128;
        float c = cosb[(size_t)s * 128 + dd], sn = sinb[(size_t)s * 128 + dd];
        out = nrm[dd] * sn + nrm[d] * c;
    }
    dst[d] = f2bf(out);
}

// ---------------- kernel 3: flash attention (softcap + causal) ----------------
// grid (NH, S/128), block 256 (8 waves); each wave owns a 16-query M-tile.
__global__ __launch_bounds__(256) void attn_kernel(
    const __bf16* __restrict__ qb, const __bf16* __restrict__ kb,
    const __bf16* __restrict__ vb, float* __restrict__ attn)
{
    extern __shared__ __align__(16) char smem[];
    __bf16* Qsw = (__bf16*)smem;             // swizzled: [8 ti][8 slab][32][16]
    __bf16* Kt  = Qsw + 128 * 256;           // [256][64] transposed K
    __bf16* Vs  = Kt + 256 * 64;             // [64][256]
    __bf16* Psw = Vs + 64 * 256;             // swizzled: [8 ti][2 slab][32][16]

    const int h    = blockIdx.x;
    const int q0   = blockIdx.y * 128;
    const int kvh  = h >> 1;
    const int t    = threadIdx.x;
    const int lane = t & 31;
    const int wid  = t >> 5;
    const int gC   = lane >> 4;
    const int nl   = lane & 15;

    // stage Q block swizzled (source already bf16; uint4 = 8 halves along d)
#pragma unroll
    for (int i = 0; i < 16; ++i) {
        int idx = i * 256 + t;
        int r = idx >> 5;            // row 0..127
        int c = (idx & 31) * 8;      // d 0..248 step 8
        uint4 q = *(const uint4*)(qb + (((size_t)(q0 + r) * NH + h) * HD) + c);
        int tile = (r >> 4) * 8 + (c >> 5);
        int L  = (r & 15) + 16 * (((c & 31) >> 3) & 1);
        int h0 = 8 * ((c >> 4) & 1);
        *(uint4*)(Qsw + ((size_t)((tile << 5) + L) << 4) + h0) = q;
    }

    v8f acc_o[16];
#pragma unroll
    for (int i = 0; i < 16; ++i) acc_o[i] = vzero8();
    float mi[8], li[8];
#pragma unroll
    for (int v = 0; v < 8; ++v) { mi[v] = -3.0e38f; li[v] = 0.0f; }

    const int kend = q0 + 128;
    for (int kb0 = 0; kb0 < kend; kb0 += 64) {
        __syncthreads();
        // stage K transposed + V straight
#pragma unroll
        for (int i = 0; i < 8; ++i) {
            int idx = i * 256 + t;
            int r = idx >> 5;
            int c = (idx & 31) * 8;
            uint4 kq = *(const uint4*)(kb + (((size_t)(kb0 + r) * NKV + kvh) * HD) + c);
            const __bf16* kh = (const __bf16*)&kq;
#pragma unroll
            for (int e = 0; e < 8; ++e) Kt[(size_t)(c + e) * 64 + r] = kh[e];
            *(uint4*)(Vs + (size_t)r * 256 + c) =
                *(const uint4*)(vb + (((size_t)(kb0 + r) * NKV + kvh) * HD) + c);
        }
        if (kb0 + 64 < kend) {  // prefetch next K/V block (global_prefetch_b8)
            __builtin_prefetch(kb + (((size_t)(kb0 + 64 + (t >> 2)) * NKV + kvh) * HD)
                                  + (t & 3) * 64, 0, 1);
            __builtin_prefetch(vb + (((size_t)(kb0 + 64 + (t >> 2)) * NKV + kvh) * HD)
                                  + (t & 3) * 64, 0, 1);
        }
        __syncthreads();

        // S = Q Kt : M=16, N=64, K=256
        v8f s4[4];
#pragma unroll
        for (int j = 0; j < 4; ++j) s4[j] = vzero8();
        for (int d0 = 0; d0 < HD; d0 += 32) {
            v16bf a = load_A_swz(Qsw, wid * 8 + (d0 >> 5), lane);
#pragma unroll
            for (int j = 0; j < 4; ++j) {
                v16bf b = load_B32x16(Kt + (size_t)d0 * 64 + 16 * j, 64, lane);
                s4[j] = wmma_bf16(a, b, s4[j]);
            }
        }

        // softcap + causal mask + online softmax
        float pv[4][8], mnew[8];
#pragma unroll
        for (int v = 0; v < 8; ++v) mnew[v] = -3.0e38f;
#pragma unroll
        for (int j = 0; j < 4; ++j)
#pragma unroll
            for (int v = 0; v < 8; ++v) {
                float sc = fast_tanh(s4[j][v] * (SCALING / SOFTCAP)) * SOFTCAP;
                int key = kb0 + 16 * j + nl;
                int row = q0 + wid * 16 + v + 8 * gC;
                if (key > row) sc = NEG_INF;
                pv[j][v] = sc;
                mnew[v] = fmaxf(mnew[v], sc);
            }
#pragma unroll
        for (int off = 1; off < 16; off <<= 1)
#pragma unroll
            for (int v = 0; v < 8; ++v)
                mnew[v] = fmaxf(mnew[v], __shfl_xor(mnew[v], off, 32));

        float alpha[8], rsum[8];
#pragma unroll
        for (int v = 0; v < 8; ++v) {
            float mn = fmaxf(mi[v], mnew[v]);
            alpha[v] = __expf(mi[v] - mn);
            mi[v] = mn;
            rsum[v] = 0.0f;
        }
        // write P into the wave-private swizzled staging area
#pragma unroll
        for (int j = 0; j < 4; ++j)
#pragma unroll
            for (int v = 0; v < 8; ++v) {
                float p = __expf(pv[j][v] - mi[v]);
                int tile = wid * 2 + (j >> 1);
                int L  = v + 8 * gC + 16 * (nl >> 3);
                int hh = (nl & 7) + 8 * (j & 1);
                Psw[((size_t)((tile << 5) + L) << 4) + hh] = f2bf(p);
                rsum[v] += p;
            }
#pragma unroll
        for (int off = 1; off < 16; off <<= 1)
#pragma unroll
            for (int v = 0; v < 8; ++v)
                rsum[v] += __shfl_xor(rsum[v], off, 32);
#pragma unroll
        for (int v = 0; v < 8; ++v) li[v] = li[v] * alpha[v] + rsum[v];
#pragma unroll
        for (int i = 0; i < 16; ++i)
#pragma unroll
            for (int v = 0; v < 8; ++v) acc_o[i][v] *= alpha[v];

        // O += P V : M=16, K=64, N=256 (Psw region is wave-private, in-wave order)
#pragma unroll
        for (int kk = 0; kk < 64; kk += 32) {
            v16bf a = load_A_swz(Psw, wid * 2 + (kk >> 5), lane);
#pragma unroll
            for (int i = 0; i < 16; ++i) {
                v16bf b = load_B32x16(Vs + (size_t)kk * 256 + 16 * i, 256, lane);
                acc_o[i] = wmma_bf16(a, b, acc_o[i]);
            }
        }
    }

    // epilogue
#pragma unroll
    for (int i = 0; i < 16; ++i)
#pragma unroll
        for (int v = 0; v < 8; ++v) {
            int row = q0 + wid * 16 + v + 8 * gC;
            int d   = 16 * i + nl;
            attn[(size_t)row * QDIM + h * HD + d] = acc_o[i][v] / li[v];
        }
}

// ---------------- kernel 4: output projection ----------------
__global__ __launch_bounds__(256) void out_gemm_kernel(
    const float* __restrict__ attn, const float* __restrict__ wo,
    float* __restrict__ out)
{
    gemm_block_128(attn, QDIM, wo, HID, out, HID, QDIM,
                   blockIdx.y * 128, blockIdx.x * 128);
}

// ---------------- host launcher ----------------
extern "C" void kernel_launch(void* const* d_in, const int* in_sizes, int n_in,
                              void* d_out, int out_size, void* d_ws, size_t ws_size,
                              hipStream_t stream)
{
    const float* hidden = (const float*)d_in[0];
    const float* cosb   = (const float*)d_in[1];
    const float* sinb   = (const float*)d_in[2];
    const float* wq = (const float*)d_in[7];
    const float* wk = (const float*)d_in[8];
    const float* wv = (const float*)d_in[9];
    const float* wo = (const float*)d_in[10];
    const float* qw = (const float*)d_in[11];
    const float* kw = (const float*)d_in[12];

    char* ws = (char*)d_ws;
    float*  xq   = (float*)(ws);                       // 2048x4096 f32
    float*  xk   = (float*)(ws + (size_t)33554432);    // 2048x2048 f32
    float*  xv   = (float*)(ws + (size_t)50331648);    // 2048x2048 f32
    __bf16* qbb  = (__bf16*)(ws + (size_t)67108864);   // 2048x4096 bf16
    __bf16* kbf  = (__bf16*)(ws + (size_t)83886080);   // 2048x2048 bf16
    __bf16* vbf  = (__bf16*)(ws + (size_t)92274688);   // 2048x2048 bf16
    float*  attn = (float*)(ws + (size_t)100663296);   // 2048x4096 f32

    qkv_gemm_kernel<<<dim3(64, 16), 256, 0, stream>>>(
        hidden, wq, wk, wv, xq, xk, xv);

    norm_rope_kernel<<<dim3(S_LEN, 32), 256, 0, stream>>>(
        xq, xk, xv, cosb, sinb, qw, kw, qbb, kbf, vbf);

    const size_t attn_lds =
        (128 * 256 + 256 * 64 + 64 * 256 + 128 * 64) * sizeof(__bf16); // 147456
    attn_kernel<<<dim3(NH, S_LEN / 128), 256, attn_lds, stream>>>(
        qbb, kbf, vbf, attn);

    out_gemm_kernel<<<dim3(24, 16), 256, 0, stream>>>(attn, wo, (float*)d_out);
}
